// RMDVAE_12833362281103
// MI455X (gfx1250) — compile-verified, matrix operand order checked by reading, not run on hardware
//
#include <hip/hip_runtime.h>
#include <hip/hip_bf16.h>

// ---------------------------------------------------------------------------
// Problem constants (match reference)
// ---------------------------------------------------------------------------
#define T_DIM   8192
#define IN_DIM  540
#define OUT_DIM 70
#define H1_DIM  256
#define H2_DIM  128
#define LAT_DIM 64
#define MCE     16
#define STD_REG 1e-3f

typedef __attribute__((ext_vector_type(16))) _Float16 v16h;
typedef __attribute__((ext_vector_type(8)))  _Float16 v8h;
typedef __attribute__((ext_vector_type(8)))  float    v8f;

// ---------------------------------------------------------------------------
// Zero-padded f32 -> f16 activation convert:  dst[M,Kpad] = pad(src[M,K])
// ---------------------------------------------------------------------------
__global__ void cvt_pad_kernel(const float* __restrict__ src, _Float16* __restrict__ dst,
                               int M, int K, int Kpad)
{
    const int i = blockIdx.x * blockDim.x + threadIdx.x;
    if (i >= M * Kpad) return;
    const int m = i / Kpad, k = i % Kpad;
    dst[i] = (k < K) ? (_Float16)src[(size_t)m * K + k] : (_Float16)0.f;
}

// ---------------------------------------------------------------------------
// Weight prep: convert+transpose into f16  Bt[Npad, Kpad], contiguous in K.
//   srcIsNK=0 : src stored [K,N]  -> Bt[n,k] = src[k,n]
//   srcIsNK=1 : src stored [N,K]  -> Bt[n,k] = src[n,k]   (gru_Wih.T case)
// ---------------------------------------------------------------------------
__global__ void prep_weight_kernel(const float* __restrict__ src, _Float16* __restrict__ dst,
                                   int K, int N, int Kpad, int Npad, int srcIsNK)
{
    const int i = blockIdx.x * blockDim.x + threadIdx.x;
    if (i >= Npad * Kpad) return;
    const int n = i / Kpad, k = i % Kpad;
    float v = 0.f;
    if (n < N && k < K)
        v = srcIsNK ? src[(size_t)n * K + k] : src[(size_t)k * N + n];
    dst[i] = (_Float16)v;
}

// ---------------------------------------------------------------------------
// WMMA GEMM:  C = epilogue(A[M,Kpad](f16) @ Bt[Npad,Kpad]^T(f16) + bias)
// One wave computes one 16x16 tile with v_wmma_f32_16x16x32_f16.
// Branch-free inner loop: per lane, A and B fragments are each two contiguous
// 16-byte chunks (documented layout: elems 0-7 -> K=kb+kHalf, 8-15 -> +16).
// out16=1: store f16 zero-padded [M,Npad] (feeds next GEMM as its A)
// out16=0: store f32 exact [M,N]
// act: 0 identity, 1 tanh, 2 exp(x)+STD_REG
// ---------------------------------------------------------------------------
__global__ __launch_bounds__(256)
void gemm_wmma_f16(const _Float16* __restrict__ A, const _Float16* __restrict__ Bt,
                   const float* __restrict__ bias,
                   float* __restrict__ Cf, _Float16* __restrict__ Ch,
                   int M, int N, int Kpad, int Npad, int act, int out16)
{
    const int lane = threadIdx.x & 31;
    const int wave = threadIdx.x >> 5;
    const int Nt = Npad >> 4;
    const int Mt = M >> 4;
    const int w  = blockIdx.x * 8 + wave;            // wave-uniform
    if (w >= Mt * Nt) return;

    const int mt = w / Nt;
    const int nt = w % Nt;
    const int mrow = (mt << 4) + (lane & 15);
    const int ncol = (nt << 4) + (lane & 15);        // < Npad always
    const int kHalf = (lane >= 16) ? 8 : 0;

    const _Float16* __restrict__ ap = A  + (size_t)mrow * Kpad + kHalf;
    const _Float16* __restrict__ bp = Bt + (size_t)ncol * Kpad + kHalf;

    v8f acc = {0.f, 0.f, 0.f, 0.f, 0.f, 0.f, 0.f, 0.f};

    for (int kb = 0; kb < Kpad; kb += 32) {
        __builtin_prefetch(ap + kb + 64, 0, 0);      // global_prefetch_b8
        const v8h a0 = *(const v8h*)(ap + kb);
        const v8h a1 = *(const v8h*)(ap + kb + 16);
        const v8h b0 = *(const v8h*)(bp + kb);
        const v8h b1 = *(const v8h*)(bp + kb + 16);
        const v16h a = __builtin_shufflevector(a0, a1, 0,1,2,3,4,5,6,7,8,9,10,11,12,13,14,15);
        const v16h b = __builtin_shufflevector(b0, b1, 0,1,2,3,4,5,6,7,8,9,10,11,12,13,14,15);
        acc = __builtin_amdgcn_wmma_f32_16x16x32_f16(
                  false, a, false, b, (short)0, acc, false, false);
    }

    // C/D layout: VGPR r -> row mt*16 + r (+8 for lanes 16-31), col = ncol
    const int rbase = (mt << 4) + ((lane >= 16) ? 8 : 0);
    const bool inN = (ncol < N);
    const float bb = inN ? bias[ncol] : 0.f;
    if (out16) {
#pragma unroll
        for (int r = 0; r < 8; ++r) {
            float v = acc[r] + bb;
            if (act == 1)      v = tanhf(v);
            else if (act == 2) v = expf(v) + STD_REG;
            Ch[(size_t)(rbase + r) * Npad + ncol] = inN ? (_Float16)v : (_Float16)0.f;
        }
    } else if (inN) {
#pragma unroll
        for (int r = 0; r < 8; ++r) {
            float v = acc[r] + bb;
            if (act == 1)      v = tanhf(v);
            else if (act == 2) v = expf(v) + STD_REG;
            Cf[(size_t)(rbase + r) * N + ncol] = v;
        }
    }
}

// ---------------------------------------------------------------------------
// Sequential GRU scan: single workgroup (192 threads = 6 waves), state in LDS.
// ---------------------------------------------------------------------------
__global__ __launch_bounds__(192)
void gru_scan_kernel(const float* __restrict__ xi,    // [T, 192]
                     const float* __restrict__ Whh,   // [192, 64]
                     const float* __restrict__ bhh,   // [192]
                     float* __restrict__ h_rnn)       // [T, 64]
{
    __shared__ float h[LAT_DIM];
    __shared__ float gh[3 * LAT_DIM];
    const int t = threadIdx.x;                        // 0..191

    float wrow[LAT_DIM];
#pragma unroll 8
    for (int j = 0; j < LAT_DIM; ++j) wrow[j] = Whh[t * LAT_DIM + j];
    const float bb = bhh[t];
    if (t < LAT_DIM) h[t] = 0.f;
    __syncthreads();

    for (int step = 0; step < T_DIM; ++step) {
        float s = bb;
#pragma unroll 8
        for (int j = 0; j < LAT_DIM; ++j) s += h[j] * wrow[j];
        gh[t] = s;
        __syncthreads();
        if (t < LAT_DIM) {
            const float xr = xi[step * 192 + t];
            const float xz = xi[step * 192 + 64 + t];
            const float xn = xi[step * 192 + 128 + t];
            const float r = 1.f / (1.f + expf(-(xr + gh[t])));
            const float z = 1.f / (1.f + expf(-(xz + gh[64 + t])));
            const float n = tanhf(xn + r * gh[128 + t]);
            const float hn = (1.f - z) * n + z * h[t];
            h[t] = hn;
            h_rnn[step * LAT_DIM + t] = hn;
        }
        __syncthreads();
    }
}

// ---------------------------------------------------------------------------
// h_alpha[t, 0:3] = h_rnn[t] @ seg_W + seg_b   (N=3: VALU)
// ---------------------------------------------------------------------------
__global__ void seg_head_kernel(const float* __restrict__ h_rnn,
                                const float* __restrict__ segW,  // [64,3]
                                const float* __restrict__ segb,  // [3]
                                float* __restrict__ h_alpha)     // [T,3]
{
    const int t = blockIdx.x * blockDim.x + threadIdx.x;
    if (t >= T_DIM) return;
    float a0 = segb[0], a1 = segb[1], a2 = segb[2];
#pragma unroll 8
    for (int j = 0; j < LAT_DIM; ++j) {
        const float hv = h_rnn[t * LAT_DIM + j];
        a0 += hv * segW[j * 3 + 0];
        a1 += hv * segW[j * 3 + 1];
        a2 += hv * segW[j * 3 + 2];
    }
    h_alpha[t * 3 + 0] = a0;
    h_alpha[t * 3 + 1] = a1;
    h_alpha[t * 3 + 2] = a2;
}

// ---------------------------------------------------------------------------
// h_mean[t,l,c] = hm_w[l,c]*h_enc[t,l] + hm_b[l,c]      (output)
// Forward value of the Gumbel-Rao straight-through sample is exactly D, so:
//   mean_mix[t,l] = h_mean[t,l,cat_idx[t]]
//   std_mix [t,l] = exp(h_logstd[l,cat_idx[t]]) + STD_REG
// ---------------------------------------------------------------------------
__global__ void mix_kernel(const _Float16* __restrict__ h_enc,  // [T,64] f16
                           const float* __restrict__ hm_w,      // [64,3]
                           const float* __restrict__ hm_b,      // [64,3]
                           const float* __restrict__ h_logstd,  // [1,64,3]
                           const int*   __restrict__ cat_idx,   // [T]
                           float* __restrict__ h_mean_out,      // [T,64,3]
                           float* __restrict__ mean_mix,        // [T,64]
                           float* __restrict__ std_mix)         // [T,64]
{
    const int i = blockIdx.x * blockDim.x + threadIdx.x;        // over T*64
    if (i >= T_DIM * LAT_DIM) return;
    const int t = i >> 6;
    const int l = i & 63;
    const float he = (float)h_enc[i];
    float m[3];
#pragma unroll
    for (int c = 0; c < 3; ++c) {
        m[c] = hm_w[l * 3 + c] * he + hm_b[l * 3 + c];
        h_mean_out[(size_t)i * 3 + c] = m[c];
    }
    const int c = cat_idx[t];
    mean_mix[i] = m[c];
    std_mix[i]  = expf(h_logstd[l * 3 + c]) + STD_REG;
}

// ---------------------------------------------------------------------------
// z16[i] = f16(mean[i] + eps[i] * std[i])    (reparameterization, feeds GEMM)
// ---------------------------------------------------------------------------
__global__ void build_z_kernel(const float* __restrict__ mean,
                               const float* __restrict__ stdv,
                               const float* __restrict__ eps,
                               _Float16* __restrict__ z, int n)
{
    const int i = blockIdx.x * blockDim.x + threadIdx.x;
    if (i < n) z[i] = (_Float16)(mean[i] + eps[i] * stdv[i]);
}

// ---------------------------------------------------------------------------
// Host orchestration
// ---------------------------------------------------------------------------
extern "C" void kernel_launch(void* const* d_in, const int* in_sizes, int n_in,
                              void* d_out, int out_size, void* d_ws, size_t ws_size,
                              hipStream_t stream)
{
    (void)in_sizes; (void)n_in; (void)out_size; (void)ws_size;

    const float* x_in    = (const float*)d_in[0];
    const float* x_out   = (const float*)d_in[1];
    const float* eps_r   = (const float*)d_in[2];
    const float* eps_h   = (const float*)d_in[3];
    /* d_in[4] = u_exp: unused — forward value of the straight-through sample is D */
    const int*   cat_idx = (const int*)  d_in[5];
    const float* enc_W0  = (const float*)d_in[6];
    const float* enc_b0  = (const float*)d_in[7];
    const float* enc_W1  = (const float*)d_in[8];
    const float* enc_b1  = (const float*)d_in[9];
    const float* enc_W2  = (const float*)d_in[10];
    const float* enc_b2  = (const float*)d_in[11];
    const float* hm_w    = (const float*)d_in[12];
    const float* hm_b    = (const float*)d_in[13];
    const float* h_logstd= (const float*)d_in[14];
    const float* gru_Wih = (const float*)d_in[15];
    const float* gru_Whh = (const float*)d_in[16];
    const float* gru_bih = (const float*)d_in[17];
    const float* gru_bhh = (const float*)d_in[18];
    const float* seg_W   = (const float*)d_in[19];
    const float* seg_b   = (const float*)d_in[20];
    const float* renc_W0 = (const float*)d_in[21];
    const float* renc_b0 = (const float*)d_in[22];
    const float* renc_W1 = (const float*)d_in[23];
    const float* renc_b1 = (const float*)d_in[24];
    const float* rmean_W = (const float*)d_in[25];
    const float* rmean_b = (const float*)d_in[26];
    const float* rstd_W  = (const float*)d_in[27];
    const float* rstd_b  = (const float*)d_in[28];
    const float* dec_W0  = (const float*)d_in[29];
    const float* dec_b0  = (const float*)d_in[30];
    const float* dec_W1  = (const float*)d_in[31];
    const float* dec_b1  = (const float*)d_in[32];
    const float* dec_W2  = (const float*)d_in[33];
    const float* dec_b2  = (const float*)d_in[34];

    // ---- output layout (flat concatenation in return order) ----
    float* out = (float*)d_out;
    float* o_hmean  = out;                 // [T,64,3]
    float* o_halpha = out + 1572864;       // [T,3]
    float* o_rmean  = out + 1597440;       // [T,64]
    float* o_rstd   = out + 2121728;       // [T,64]
    float* o_routr  = out + 2646016;       // [16,T,70]
    float* o_routh  = out + 11821056;      // [16,T,70]

    // ---- workspace layout ----
    float* ws       = (float*)d_ws;
    float* xi       = ws;                  // [T,192]  1,572,864 f
    float* h_rnn    = ws + 1572864;        // [T,64]     524,288 f
    float* mean_mix = ws + 2097152;        // [T,64]     524,288 f
    float* std_mix  = ws + 2621440;        // [T,64]     524,288 f
    _Float16* hbase = (_Float16*)(ws + 3145728);
    _Float16* x_in16  = hbase;             // [T,544]  4,456,448 h
    _Float16* x_out16 = x_in16  + 4456448; // [T,96]     786,432 h
    _Float16* bufA16  = x_out16 + 786432;  // [T,256]  2,097,152 h  (A1 / R1)
    _Float16* bufB16  = bufA16  + 2097152; // [T,128]  1,048,576 h  (A2 / r_enc)
    _Float16* h_enc16 = bufB16  + 1048576; // [T,64]     524,288 h
    _Float16* z16     = h_enc16 + 524288;  // [T,64]     524,288 h
    _Float16* d116    = z16     + 524288;  // [T,128]  1,048,576 h
    _Float16* d216    = d116    + 1048576; // [T,256]  2,097,152 h
    _Float16* wb      = d216    + 2097152; // weight area
    _Float16* enc_W0t = wb;                // 256x544
    _Float16* enc_W1t = enc_W0t + 139264;  // 128x256
    _Float16* enc_W2t = enc_W1t + 32768;   //  64x128
    _Float16* gru_Wt  = enc_W2t + 8192;    // 192x64
    _Float16* renc_W0t= gru_Wt  + 12288;   // 256x96
    _Float16* renc_W1t= renc_W0t+ 24576;   // 128x256
    _Float16* rmean_Wt= renc_W1t+ 32768;   //  64x128
    _Float16* rstd_Wt = rmean_Wt+ 8192;    //  64x128
    _Float16* dec_W0t = rstd_Wt + 8192;    // 128x64
    _Float16* dec_W1t = dec_W0t + 8192;    // 256x128
    _Float16* dec_W2t = dec_W1t + 32768;   //  80x256

    auto kpad32 = [](int k) { return (k + 31) & ~31; };

    auto prepW = [&](const float* src, _Float16* dst, int K, int N, int srcIsNK) {
        const int Kp = kpad32(K), Np = (N + 15) & ~15;
        prep_weight_kernel<<<(Np * Kp + 255) / 256, 256, 0, stream>>>(src, dst, K, N, Kp, Np, srcIsNK);
    };
    auto cvtA = [&](const float* src, _Float16* dst, int M, int K) {
        const int Kp = kpad32(K);
        cvt_pad_kernel<<<(M * Kp + 255) / 256, 256, 0, stream>>>(src, dst, M, K, Kp);
    };
    auto gemm = [&](const _Float16* A, const _Float16* Bt, const float* bias,
                    float* Cf, _Float16* Ch, int M, int N, int K, int act) {
        const int Kp = kpad32(K);
        const int out16 = (Ch != nullptr);
        const int Np = out16 ? kpad32(N) : ((N + 15) & ~15);  // f16 pad matches next Kpad
        const int blocks = ((M >> 4) * (Np >> 4) + 7) / 8;    // 8 waves / 256-thr block
        gemm_wmma_f16<<<blocks, 256, 0, stream>>>(A, Bt, bias, Cf, Ch, M, N, Kp, Np, act, out16);
    };

    // ---- one-time f16 prep (weights transposed to [Npad,Kpad], inputs padded) ----
    prepW(enc_W0,  enc_W0t,  IN_DIM, H1_DIM, 0);
    prepW(enc_W1,  enc_W1t,  H1_DIM, H2_DIM, 0);
    prepW(enc_W2,  enc_W2t,  H2_DIM, LAT_DIM, 0);
    prepW(gru_Wih, gru_Wt,   LAT_DIM, 3 * LAT_DIM, 1);   // already [N,K]
    prepW(renc_W0, renc_W0t, OUT_DIM, H1_DIM, 0);
    prepW(renc_W1, renc_W1t, H1_DIM, H2_DIM, 0);
    prepW(rmean_W, rmean_Wt, H2_DIM, LAT_DIM, 0);
    prepW(rstd_W,  rstd_Wt,  H2_DIM, LAT_DIM, 0);
    prepW(dec_W0,  dec_W0t,  LAT_DIM, H2_DIM, 0);
    prepW(dec_W1,  dec_W1t,  H2_DIM, H1_DIM, 0);
    prepW(dec_W2,  dec_W2t,  H1_DIM, OUT_DIM, 0);
    cvtA(x_in,  x_in16,  T_DIM, IN_DIM);
    cvtA(x_out, x_out16, T_DIM, OUT_DIM);

    // ---- human encoder: 540 -> 256 -> 128 -> 64 (tanh) ----
    gemm(x_in16, enc_W0t, enc_b0, nullptr, bufA16,  T_DIM, H1_DIM, IN_DIM, 1);
    gemm(bufA16, enc_W1t, enc_b1, nullptr, bufB16,  T_DIM, H2_DIM, H1_DIM, 1);
    gemm(bufB16, enc_W2t, enc_b2, nullptr, h_enc16, T_DIM, LAT_DIM, H2_DIM, 1);

    // ---- GRU: xi = h_enc @ Wih.T + bih (f32), then sequential scan ----
    gemm(h_enc16, gru_Wt, gru_bih, xi, nullptr, T_DIM, 3 * LAT_DIM, LAT_DIM, 0);
    gru_scan_kernel<<<1, 192, 0, stream>>>(xi, gru_Whh, gru_bhh, h_rnn);
    seg_head_kernel<<<(T_DIM + 255) / 256, 256, 0, stream>>>(h_rnn, seg_W, seg_b, o_halpha);

    // ---- mixture means + category-selected mean/std (forward sample = D) ----
    mix_kernel<<<(T_DIM * LAT_DIM + 255) / 256, 256, 0, stream>>>(
        h_enc16, hm_w, hm_b, h_logstd, cat_idx, o_hmean, mean_mix, std_mix);

    // ---- robot encoder + heads ----
    gemm(x_out16, renc_W0t, renc_b0, nullptr, bufA16, T_DIM, H1_DIM, OUT_DIM, 1);
    gemm(bufA16,  renc_W1t, renc_b1, nullptr, bufB16, T_DIM, H2_DIM, H1_DIM, 1);
    gemm(bufB16,  rmean_Wt, rmean_b, o_rmean, nullptr, T_DIM, LAT_DIM, H2_DIM, 1);
    gemm(bufB16,  rstd_Wt,  rstd_b,  o_rstd,  nullptr, T_DIM, LAT_DIM, H2_DIM, 2);

    // ---- decoder, chunked per MCE sample: 64 -> 128 -> 256 -> 70 (tanh) ----
    const int TL = T_DIM * LAT_DIM;
    for (int s = 0; s < MCE; ++s) {          // robot-branch samples
        build_z_kernel<<<(TL + 255) / 256, 256, 0, stream>>>(
            o_rmean, o_rstd, eps_r + (size_t)s * TL, z16, TL);
        gemm(z16,  dec_W0t, dec_b0, nullptr, d116, T_DIM, H2_DIM, LAT_DIM, 1);
        gemm(d116, dec_W1t, dec_b1, nullptr, d216, T_DIM, H1_DIM, H2_DIM, 1);
        gemm(d216, dec_W2t, dec_b2, o_routr + (size_t)s * T_DIM * OUT_DIM, nullptr,
             T_DIM, OUT_DIM, H1_DIM, 1);
    }
    for (int s = 0; s < MCE; ++s) {          // human-branch samples
        build_z_kernel<<<(TL + 255) / 256, 256, 0, stream>>>(
            mean_mix, std_mix, eps_h + (size_t)s * TL, z16, TL);
        gemm(z16,  dec_W0t, dec_b0, nullptr, d116, T_DIM, H2_DIM, LAT_DIM, 1);
        gemm(d116, dec_W1t, dec_b1, nullptr, d216, T_DIM, H1_DIM, H2_DIM, 1);
        gemm(d216, dec_W2t, dec_b2, o_routh + (size_t)s * T_DIM * OUT_DIM, nullptr,
             T_DIM, OUT_DIM, H1_DIM, 1);
    }
}
